// SeasonalFNO1D_1434519077292
// MI455X (gfx1250) — matrix operand order repaired and checked by reading, not compiled, not run on hardware
//
#include <hip/hip_runtime.h>
#include <hip/hip_bf16.h>
#include <math.h>

#define B_ 64
#define T_ 8192
#define F_ 6
#define MODES_ 32
#define W_ 64
#define LAYERS_ 4
#define FC1_ 128
#define OUT_ 7
#define LDK 72      // padded LDS row stride (bf16 elems); 144B rows, 16B aligned
#define KSPLIT 8    // K-dimension split of the forward DFT GEMM

typedef __bf16 bf16;
typedef __attribute__((ext_vector_type(16))) __bf16 v16bf;
typedef __attribute__((ext_vector_type(8)))  float  v8f;

__device__ __forceinline__ float gelu_f(float v) {
  return 0.5f * v * (1.0f + erff(v * 0.70710678118654752f));
}

// ---------------- WMMA fragment helpers (wave32, v_wmma_f32_16x16x32_bf16) -------------
// A-matrix 16x32 (MxK), LDS layout [row][k] row-major, stride LDK.
// lane: m = lane&15, h = lane>>4; lane holds K = 16h .. 16h+15 (contiguous, matches B).
__device__ __forceinline__ v16bf load_frag_a(const bf16* lds, int row0, int k0, int lane) {
  const int m = lane & 15, h = lane >> 4;
  const bf16* p = lds + (row0 + m) * LDK + k0;
  v16bf f;
#pragma unroll
  for (int v = 0; v < 16; ++v) f[v] = p[16*h + v];
  return f;
}

// B-matrix 32x16 (KxN), staged TRANSPOSED in LDS as [n][k], stride LDK.
// lane: n = lane&15, h = lane>>4; lane holds K = 16h .. 16h+15 (contiguous -> b128 loads).
__device__ __forceinline__ v16bf load_frag_b(const bf16* lds, int n0, int k0, int lane) {
  const int n = lane & 15, h = lane >> 4;
  const bf16* p = lds + (n0 + n) * LDK + k0 + 16 * h;
  v16bf f;
#pragma unroll
  for (int v = 0; v < 16; ++v) f[v] = p[v];
  return f;
}

__device__ __forceinline__ v8f wmma_bf16(v16bf a, v16bf b, v8f c) {
  // (neg_a, A, neg_b, B, c_mod, C, reuse_a, reuse_b)
  return __builtin_amdgcn_wmma_f32_16x16x32_bf16(false, a, false, b, (short)0, c, false, false);
}

// ---------------- trig tables: dftA[64][T] (rows: 32 cos, 32 -sin),
// idftW[T][64] (cols: ck*cos/T, -2*sin/T; col 32 (DC imag) = 0) ----------------
__global__ void k_trig(bf16* __restrict__ dftA, bf16* __restrict__ idftW) {
  int idx = blockIdx.x * blockDim.x + threadIdx.x;
  const float w0 = 6.283185307179586f / (float)T_;
  if (idx < 64 * T_) {
    int k = idx / T_, t = idx - k * T_;
    float v;
    if (k < MODES_) {
      int r = (k * t) & (T_ - 1);
      v = cosf((float)r * w0);
    } else {
      int kk = k - MODES_;
      int r = (kk * t) & (T_ - 1);
      v = -sinf((float)r * w0);
    }
    dftA[idx] = (bf16)v;
  } else {
    idx -= 64 * T_;
    int t = idx >> 6, k = idx & 63;
    float v;
    if (k < MODES_) {
      int r = (k * t) & (T_ - 1);
      float c = (k == 0) ? 1.0f : 2.0f;
      v = c * (1.0f / (float)T_) * cosf((float)r * w0);
    } else {
      int kk = k - MODES_;
      if (kk == 0) v = 0.0f;   // irfft ignores DC imaginary part
      else {
        int r = (kk * t) & (T_ - 1);
        v = -2.0f * (1.0f / (float)T_) * sinf((float)r * w0);
      }
    }
    idftW[(size_t)t * 64 + k] = (bf16)v;
  }
}

// ---------------- fc0 lift: x[b,t,c] = inputs[b,t,:6] @ fc0_w + fc0_b ----------------
__global__ void k_fc0(const float* __restrict__ in, const float* __restrict__ w,
                      const float* __restrict__ b, bf16* __restrict__ x) {
  size_t idx = (size_t)blockIdx.x * blockDim.x + threadIdx.x;  // B*T*W
  int c = (int)(idx & 63);
  size_t bt = idx >> 6;
  float acc = b[c];
#pragma unroll
  for (int f = 0; f < F_; ++f) acc += in[bt * F_ + f] * w[f * W_ + c];
  x[idx] = (bf16)acc;
}

// ---------------- forward 32-mode DFT as GEMM (K split 8-way, double-buffered LDS):
// modesP[split][b] = dftA[:, kbeg:kend] @ x_b[kbeg:kend, :]
__global__ void __launch_bounds__(512) k_dft(const bf16* __restrict__ dftA,
                                             const bf16* __restrict__ x,
                                             float* __restrict__ modesP) {
  __shared__ bf16 lds_a[2][64 * LDK];   // dftA chunk  [m][k]
  __shared__ bf16 lds_x[2][64 * LDK];   // x chunk transposed [c][k]
  const int b = blockIdx.x, split = blockIdx.y;
  const int tid = threadIdx.x;
  const int lane = tid & 31, wave = tid >> 5;     // 16 waves, one 16x16 tile each
  const int mi = wave >> 2, ni = wave & 3;
  const bf16* xb = x + (size_t)b * T_ * W_;
  const int kbeg = split * (T_ / KSPLIT);
  const int iters = (T_ / KSPLIT) / 64;           // 16

  // staging index precompute
  const int arow = tid >> 3, acol = (tid & 7) * 8;   // A copy: 8 contiguous elems
  const int kk2 = (tid >> 4) * 2, c0 = (tid & 15) * 4; // X: 2 k-rows x 4 channels

  auto stage = [&](int buf, int k0) {
    // dftA chunk: straight b128 copy
    *reinterpret_cast<uint4*>(&lds_a[buf][arow * LDK + acol]) =
        *reinterpret_cast<const uint4*>(dftA + (size_t)arow * T_ + k0 + acol);
    // x chunk: transpose with k-pairs packed into dwords -> ds_store_b32 x4
    union { uint2 v; unsigned short s[4]; } r0, r1;
    r0.v = *reinterpret_cast<const uint2*>(xb + (size_t)(k0 + kk2) * W_ + c0);
    r1.v = *reinterpret_cast<const uint2*>(xb + (size_t)(k0 + kk2 + 1) * W_ + c0);
#pragma unroll
    for (int j = 0; j < 4; ++j) {
      unsigned int packed = (unsigned int)r0.s[j] | ((unsigned int)r1.s[j] << 16);
      *reinterpret_cast<unsigned int*>(&lds_x[buf][(c0 + j) * LDK + kk2]) = packed;
    }
  };

  v8f acc = {};
  stage(0, kbeg);
  __syncthreads();
  for (int i = 0; i < iters; ++i) {
    if (i + 1 < iters) stage((i + 1) & 1, kbeg + (i + 1) * 64);
    const bf16* la = lds_a[i & 1];
    const bf16* lx = lds_x[i & 1];
#pragma unroll
    for (int kk = 0; kk < 64; kk += 32) {
      v16bf a  = load_frag_a(la, mi * 16, kk, lane);
      v16bf bb = load_frag_b(lx, ni * 16, kk, lane);
      acc = wmma_bf16(a, bb, acc);
    }
    __syncthreads();
  }
  float* out = modesP + ((size_t)split * B_ + b) * 4096;
  const int n = lane & 15, h = lane >> 4;
#pragma unroll
  for (int v = 0; v < 8; ++v)
    out[(mi * 16 + v + 8 * h) * 64 + ni * 16 + n] = acc[v];
}

// ---------------- deterministic K-split reduction: modesF = sum_p modesP[p] ----------------
__global__ void k_reduce(const float* __restrict__ modesP, float* __restrict__ modesF) {
  int idx = blockIdx.x * blockDim.x + threadIdx.x;   // B*64*64
  float s = 0.f;
#pragma unroll
  for (int p = 0; p < KSPLIT; ++p) s += modesP[(size_t)p * B_ * 4096 + idx];
  modesF[idx] = s;
}

// ---------------- complex mode mixing: out_ft[b,k,o] = sum_c x_ft[b,k,c] * (wr+i*wi)[k,c,o]
// stores result TRANSPOSED as mixed[b][o][r] bf16 (r<32: Re, r>=32: Im) for IDFT B-operand.
__global__ void k_mix(const float* __restrict__ modesF, const float* __restrict__ wr,
                      const float* __restrict__ wi, bf16* __restrict__ mixed) {
  int idx = blockIdx.x * blockDim.x + threadIdx.x;  // B*MODES*W
  int o = idx & 63;
  int k = (idx >> 6) & 31;
  int b = idx >> 11;
  const float* re = modesF + ((size_t)b * 64 + k) * 64;
  const float* im = modesF + ((size_t)b * 64 + 32 + k) * 64;
  const float* wrk = wr + (size_t)k * 64 * 64 + o;
  const float* wik = wi + (size_t)k * 64 * 64 + o;
  float aR = 0.f, aI = 0.f;
#pragma unroll 4
  for (int c = 0; c < 64; ++c) {
    float r = re[c], i = im[c];
    float pr = wrk[(size_t)c * 64], pi = wik[(size_t)c * 64];
    aR += r * pr - i * pi;
    aI += r * pi + i * pr;
  }
  bf16* mb = mixed + ((size_t)b * 64 + o) * 64;
  mb[k]      = (bf16)aR;
  mb[32 + k] = (bf16)aI;
}

// ---------------- fused IDFT + pointwise conv + bias + GELU, in-place on x --------------
// x_new[t,:] = gelu( idftW[t,:] @ mixed_b  +  x[t,:] @ conv_w  + conv_b )
// 512 threads, 128 output rows per block (amortizes B-operand staging), 16 waves x 2 tiles.
__global__ void __launch_bounds__(512) k_idft_conv(const bf16* __restrict__ idftW,
                                                   const bf16* __restrict__ mixed,
                                                   const float* __restrict__ convw,
                                                   const float* __restrict__ convb,
                                                   bf16* __restrict__ x) {
  __shared__ bf16 a1[128 * LDK];  // idftW rows t0..t0+127  [t][r]
  __shared__ bf16 a2[128 * LDK];  // x rows t0..t0+127      [t][c]
  __shared__ bf16 b1[64 * LDK];   // mixed_b (pre-transposed) [o][r]
  __shared__ bf16 b2[64 * LDK];   // conv_w transposed        [o][c]
  const int b  = blockIdx.x;
  const int t0 = blockIdx.y * 128;
  const int tid = threadIdx.x;
  {   // a1/a2: 16 elems/thread (2 x uint4)
    int row = tid >> 2, col = (tid & 3) * 16;
    const bf16* s1 = idftW + (size_t)(t0 + row) * 64 + col;
    const bf16* s2 = x + (((size_t)b * T_) + t0 + row) * W_ + col;
    *reinterpret_cast<uint4*>(&a1[row * LDK + col])     = *reinterpret_cast<const uint4*>(s1);
    *reinterpret_cast<uint4*>(&a1[row * LDK + col + 8]) = *reinterpret_cast<const uint4*>(s1 + 8);
    *reinterpret_cast<uint4*>(&a2[row * LDK + col])     = *reinterpret_cast<const uint4*>(s2);
    *reinterpret_cast<uint4*>(&a2[row * LDK + col + 8]) = *reinterpret_cast<const uint4*>(s2 + 8);
  }
  {   // b1: direct b128 copy (already [o][r] in global)
    int row = tid >> 3, col = (tid & 7) * 8;
    *reinterpret_cast<uint4*>(&b1[row * LDK + col]) =
        *reinterpret_cast<const uint4*>(mixed + (size_t)b * 64 * 64 + (size_t)row * 64 + col);
  }
#pragma unroll
  for (int j = 0; j < 8; ++j) {   // b2: conv_w f32 -> bf16, transposed
    int idx = tid * 8 + j;
    int c = idx >> 6, o = idx & 63;
    b2[o * LDK + c] = (bf16)convw[(size_t)c * 64 + o];
  }
  __syncthreads();
  const int lane = tid & 31, wave = tid >> 5;   // 16 waves, 2 tiles each (8x4 tile grid)
#pragma unroll
  for (int s = 0; s < 2; ++s) {
    int tile = wave * 2 + s;
    int mi = tile >> 2, ni = tile & 3;          // mi in 0..7, ni in 0..3
    v8f acc = {};
#pragma unroll
    for (int kk = 0; kk < 64; kk += 32) {
      acc = wmma_bf16(load_frag_a(a1, mi * 16, kk, lane), load_frag_b(b1, ni * 16, kk, lane), acc);
      acc = wmma_bf16(load_frag_a(a2, mi * 16, kk, lane), load_frag_b(b2, ni * 16, kk, lane), acc);
    }
    const int n = lane & 15, h = lane >> 4;
    const int cc = ni * 16 + n;
    const float bias = convb[cc];
#pragma unroll
    for (int v = 0; v < 8; ++v) {
      int m = mi * 16 + v + 8 * h;
      x[((size_t)b * T_ + t0 + m) * W_ + cc] = (bf16)gelu_f(acc[v] + bias);
    }
  }
}

// ---------------- head: only t = T-1 survives the slice ----------------
__global__ void k_head(const bf16* __restrict__ x, const float* __restrict__ w1,
                       const float* __restrict__ b1, const float* __restrict__ w2,
                       const float* __restrict__ b2, float* __restrict__ out) {
  __shared__ float hh[FC1_];
  __shared__ float xs[W_];
  const int b = blockIdx.x;
  const int tid = threadIdx.x;
  if (tid < W_) xs[tid] = (float)x[((size_t)b * T_ + T_ - 1) * W_ + tid];
  __syncthreads();
  float acc = b1[tid];
#pragma unroll 8
  for (int c = 0; c < W_; ++c) acc += xs[c] * w1[(size_t)c * FC1_ + tid];
  hh[tid] = gelu_f(acc);
  __syncthreads();
  if (tid < OUT_) {
    float a = b2[tid];
#pragma unroll 8
    for (int k = 0; k < FC1_; ++k) a += hh[k] * w2[(size_t)k * OUT_ + tid];
    out[(size_t)b * OUT_ + tid] = a;
  }
}

extern "C" void kernel_launch(void* const* d_in, const int* in_sizes, int n_in,
                              void* d_out, int out_size, void* d_ws, size_t ws_size,
                              hipStream_t stream) {
  (void)in_sizes; (void)n_in; (void)out_size; (void)ws_size;
  const float* inputs = (const float*)d_in[0];
  const float* fc0_w  = (const float*)d_in[1];
  const float* fc0_b  = (const float*)d_in[2];
  const float* fwr    = (const float*)d_in[3];
  const float* fwi    = (const float*)d_in[4];
  const float* convw  = (const float*)d_in[5];
  const float* convb  = (const float*)d_in[6];
  const float* fc1_w  = (const float*)d_in[7];
  const float* fc1_b  = (const float*)d_in[8];
  const float* fc2_w  = (const float*)d_in[9];
  const float* fc2_b  = (const float*)d_in[10];
  float* out = (float*)d_out;

  char* ws = (char*)d_ws;
  auto carve = [&](size_t bytes) -> char* {
    char* p = ws; ws += (bytes + 255) & ~(size_t)255; return p;
  };
  bf16*  x      = (bf16*) carve((size_t)B_ * T_ * W_ * sizeof(bf16));            // 64 MB
  bf16*  dftA   = (bf16*) carve((size_t)64 * T_ * sizeof(bf16));                 // 1 MB
  bf16*  idftW  = (bf16*) carve((size_t)T_ * 64 * sizeof(bf16));                 // 1 MB
  float* modesP = (float*)carve((size_t)KSPLIT * B_ * 64 * 64 * sizeof(float));  // 8 MB
  float* modesF = (float*)carve((size_t)B_ * 64 * 64 * sizeof(float));           // 1 MB
  bf16*  mixed  = (bf16*) carve((size_t)B_ * 64 * 64 * sizeof(bf16));            // 0.5 MB

  k_trig<<<(2 * 64 * T_) / 256, 256, 0, stream>>>(dftA, idftW);
  k_fc0 <<<(B_ * T_ * W_) / 256, 256, 0, stream>>>(inputs, fc0_w, fc0_b, x);
  for (int l = 0; l < LAYERS_; ++l) {
    k_dft<<<dim3(B_, KSPLIT), 512, 0, stream>>>(dftA, x, modesP);
    k_reduce<<<(B_ * 64 * 64) / 256, 256, 0, stream>>>(modesP, modesF);
    k_mix<<<(B_ * MODES_ * W_) / 256, 256, 0, stream>>>(
        modesF, fwr + (size_t)l * MODES_ * W_ * W_, fwi + (size_t)l * MODES_ * W_ * W_, mixed);
    k_idft_conv<<<dim3(B_, T_ / 128), 512, 0, stream>>>(
        idftW, mixed, convw + (size_t)l * W_ * W_, convb + (size_t)l * W_, x);
  }
  k_head<<<B_, FC1_, 0, stream>>>(x, fc1_w, fc1_b, fc2_w, fc2_b, out);
}
